// Viterbi_23210003267712
// MI455X (gfx1250) — compile-verified, matrix operand order checked by reading, not run on hardware
//
#include <hip/hip_runtime.h>
#include <hip/hip_fp16.h>

#define TT 128
#define BB 8
#define SS 512
#define NWG_PER_B 32        // 512 rows / 16 rows per WG
#define ROWS_PER_WG 16
#define THREADS 256         // 8 waves (wave32)
#define NWAVES 8

typedef __attribute__((ext_vector_type(16))) _Float16 v16h;
typedef __attribute__((ext_vector_type(8)))  float    v8f;

__global__ void vit_init_counters(unsigned* cnt) {
    if (threadIdx.x < BB) cnt[threadIdx.x] = 0u;
}

// One chain (batch b) is handled by 32 workgroups; 128 sequential steps with a
// per-chain global arrival barrier between steps. V buffers ping-pong in d_ws.
__global__ __launch_bounds__(THREADS)
void vit_chain_kernel(const float* __restrict__ theta,
                      float* __restrict__ vbuf,     // [2][BB][SS] in d_ws
                      unsigned* __restrict__ cnt)   // [BB] arrival counters
{
    __shared__ __align__(16) float vs[SS];              // V[j] - max(V)
    __shared__ float red[THREADS];
    __shared__ float partial[ROWS_PER_WG * NWAVES];

    const int tid  = threadIdx.x;
    const int lane = tid & 31;
    const int wave = tid >> 5;
    const int b    = blockIdx.x >> 5;                   // chain id
    const int row0 = (blockIdx.x & 31) * ROWS_PER_WG;   // output rows of this WG
    const int m    = lane & 15;                         // A-fragment row within tile
    const int sel  = lane >> 4;                         // lane-group (K half) select

    v16h ones;
    #pragma unroll
    for (int q = 0; q < 16; ++q) ones[q] = (_Float16)1.0f;

    unsigned* mycnt = cnt + b;

    for (int t = 0; t < TT; ++t) {
        const float* Vin  = vbuf + (size_t)(t & 1) * BB * SS + (size_t)b * SS;
        float*       Vout = vbuf + (size_t)((t + 1) & 1) * BB * SS + (size_t)b * SS;

        // ---- M = max_j V[j]; vs[j] = V[j] - M (t==0: V == 0) ----
        float M = 0.0f;
        if (t > 0) {
            float a0 = Vin[tid];
            float a1 = Vin[tid + THREADS];
            red[tid] = fmaxf(a0, a1);
            __syncthreads();
            for (int s = THREADS / 2; s > 0; s >>= 1) {
                if (tid < s) red[tid] = fmaxf(red[tid], red[tid + s]);
                __syncthreads();
            }
            M = red[0];
            vs[tid]           = a0 - M;
            vs[tid + THREADS] = a1 - M;
        } else {
            vs[tid]           = 0.0f;
            vs[tid + THREADS] = 0.0f;
        }
        __syncthreads();

        // ---- row sums of exp(theta + vs) via WMMA with all-ones B ----
        // A fragment (16-bit A 16x32 layout): lanes 0-15 hold K in {0..7,16..23},
        // lanes 16-31 hold K in {8..15,24..31}; element e -> VGPR e/2, half e%2.
        const float* rowp =
            theta + ((((size_t)t * BB + b) * SS) + (size_t)(row0 + m)) * SS;
        v8f acc = {};
        #pragma unroll
        for (int cc = 0; cc < 2; ++cc) {
            const int k0 = (wave * 2 + cc) * 32;   // wave covers K = [64w, 64w+64)
            const int kA = k0 + sel * 8;
            const int kB = kA + 16;
            float4 q0 = *(const float4*)(rowp + kA);
            float4 q1 = *(const float4*)(rowp + kA + 4);
            float4 q2 = *(const float4*)(rowp + kB);
            float4 q3 = *(const float4*)(rowp + kB + 4);
            float4 u0 = *(const float4*)(vs + kA);
            float4 u1 = *(const float4*)(vs + kA + 4);
            float4 u2 = *(const float4*)(vs + kB);
            float4 u3 = *(const float4*)(vs + kB + 4);
            float tf[16] = {q0.x,q0.y,q0.z,q0.w, q1.x,q1.y,q1.z,q1.w,
                            q2.x,q2.y,q2.z,q2.w, q3.x,q3.y,q3.z,q3.w};
            float vf[16] = {u0.x,u0.y,u0.z,u0.w, u1.x,u1.y,u1.z,u1.w,
                            u2.x,u2.y,u2.z,u2.w, u3.x,u3.y,u3.z,u3.w};
            v16h aHi, aLo;
            #pragma unroll
            for (int q = 0; q < 16; ++q) {
                float p = __expf(tf[q] + vf[q]);   // <= exp(~6): fits f16
                _Float16 ph = (_Float16)p;
                aHi[q] = ph;
                aLo[q] = (_Float16)(p - (float)ph); // Dekker residual -> ~f32 sum
            }
            acc = __builtin_amdgcn_wmma_f32_16x16x32_f16(
                      false, aHi, false, ones, (short)0, acc, false, false);
            acc = __builtin_amdgcn_wmma_f32_16x16x32_f16(
                      false, aLo, false, ones, (short)0, acc, false, false);
        }

        // prefetch next timestep's tile (same rows, +B*S*S floats)
        if (t + 1 < TT) {
            __builtin_prefetch(rowp + (size_t)BB * SS * SS + (wave * 64), 0, 1);
        }

        // C/D layout: VGPR r, lanes 0-15 -> M=r; lanes 16-31 -> M=8+r.
        // All N columns equal the row sum (B == ones); take column 0 per group.
        if ((lane & 15) == 0) {
            #pragma unroll
            for (int r = 0; r < 8; ++r)
                partial[(sel * 8 + r) * NWAVES + wave] = acc[r];
        }
        __syncthreads();

        if (tid < ROWS_PER_WG) {
            float s = 0.0f;
            #pragma unroll
            for (int w = 0; w < NWAVES; ++w) s += partial[tid * NWAVES + w];
            Vout[row0 + tid] = M + __logf(s);
        }

        // ---- per-chain global barrier (release / arrive / spin / acquire) ----
        __threadfence();
        __syncthreads();
        if (tid == 0) {
            atomicAdd(mycnt, 1u);
            const unsigned target = (unsigned)(NWG_PER_B * (t + 1));
            while (__hip_atomic_load(mycnt, __ATOMIC_RELAXED,
                                     __HIP_MEMORY_SCOPE_AGENT) < target) {
                __builtin_amdgcn_s_sleep(2);
            }
        }
        __syncthreads();
        __threadfence();
    }
}

// out[b] = logsumexp_j V_final[b, j]; final V lives in buffer (TT & 1) == 0.
__global__ __launch_bounds__(256)
void vit_final_lse(const float* __restrict__ vbuf, float* __restrict__ out) {
    __shared__ float red[256];
    const int tid = threadIdx.x;
    const int b   = blockIdx.x;
    const float* V = vbuf + (size_t)(TT & 1) * BB * SS + (size_t)b * SS;
    float a0 = V[tid], a1 = V[tid + 256];
    red[tid] = fmaxf(a0, a1);
    __syncthreads();
    for (int s = 128; s > 0; s >>= 1) {
        if (tid < s) red[tid] = fmaxf(red[tid], red[tid + s]);
        __syncthreads();
    }
    float M = red[0];
    __syncthreads();
    red[tid] = __expf(a0 - M) + __expf(a1 - M);
    __syncthreads();
    for (int s = 128; s > 0; s >>= 1) {
        if (tid < s) red[tid] += red[tid + s];
        __syncthreads();
    }
    if (tid == 0) out[b] = M + __logf(red[0]);
}

extern "C" void kernel_launch(void* const* d_in, const int* in_sizes, int n_in,
                              void* d_out, int out_size, void* d_ws, size_t ws_size,
                              hipStream_t stream) {
    const float* theta = (const float*)d_in[0];
    float* out  = (float*)d_out;
    float* vbuf = (float*)d_ws;                                  // 2*8*512 f32 = 32 KiB
    unsigned* cnt = (unsigned*)((char*)d_ws + 2 * BB * SS * sizeof(float));
    (void)in_sizes; (void)n_in; (void)out_size; (void)ws_size;

    vit_init_counters<<<1, 32, 0, stream>>>(cnt);
    vit_chain_kernel<<<BB * NWG_PER_B, THREADS, 0, stream>>>(theta, vbuf, cnt);
    vit_final_lse<<<BB, 256, 0, stream>>>(vbuf, out);
}
